// SparseMOE_75161927680704
// MI455X (gfx1250) — matrix operand order repaired and checked by reading, not compiled, not run on hardware
//
#include <hip/hip_runtime.h>

// ---------------- problem constants ----------------
#define Bb 4
#define Ss 2048
#define Dd 2048
#define Hh 2048
#define Oo 2048
#define Ee 8
#define Kk 2
#define NTOK (Bb * Ss)        // 8192
#define CAP 2560              // int(1.25 * NTOK * Kk / Ee)

// ---------------- GEMM tiling ----------------
#define BM 128
#define BN 128
#define BKE 64                 // k-elems per LDS stage (2 WMMA substeps of 32)
#define A_ELEMS (BM * BKE)     // 8192
#define B_ELEMS (BN * BKE)     // 8192
#define TILE_ELEMS (A_ELEMS + B_ELEMS)   // 16384 elems = 32 KB
#define BUF_BYTES (TILE_ELEMS * 2)

typedef __attribute__((ext_vector_type(16))) __bf16 bf16x16;
typedef __attribute__((ext_vector_type(8)))  __bf16 bf16x8;
typedef __attribute__((ext_vector_type(8)))  float  f32x8;

union AFrag { bf16x16 v; bf16x8 h[2]; };

static __device__ __forceinline__ __bf16 to_bf16(float f) { return (__bf16)f; }

// async global -> LDS DMA (ASYNCcnt path). VDST = LDS byte address.
template <int OFF>
static __device__ __forceinline__ void async_copy_b128(unsigned lds_addr, const void* gaddr) {
    asm volatile("global_load_async_to_lds_b128 %0, %1, off offset:%c2"
                 :: "v"(lds_addr), "v"(gaddr), "n"(OFF)
                 : "memory");
}
static __device__ __forceinline__ void wait_async8() {
    asm volatile("s_wait_asynccnt 0x8" ::: "memory");
}
static __device__ __forceinline__ void wait_async0() {
    asm volatile("s_wait_asynccnt 0x0" ::: "memory");
}

// ---------------- init kernels ----------------
__global__ void init_meta_kernel(int* cnt, float* psum, int* tok, float* wgt) {
    int i = blockIdx.x * blockDim.x + threadIdx.x;
    if (i < Ee) { cnt[i] = 0; psum[i] = 0.0f; }
    if (i < Ee * CAP) { tok[i] = 0; wgt[i] = 0.0f; }
}

__global__ void out_init_kernel(float4* out, const float4* x, int n4) {
    int i = blockIdx.x * blockDim.x + threadIdx.x;
    int stride = gridDim.x * blockDim.x;
    for (; i < n4; i += stride) out[i] = x[i];
}

__global__ void cvt_x_kernel(const float* x, __bf16* xb, int n) {
    int i = blockIdx.x * blockDim.x + threadIdx.x;
    int stride = gridDim.x * blockDim.x;
    for (; i < n; i += stride) xb[i] = to_bf16(x[i]);
}

// in: [z][R][C] f32  ->  out: [z][C][R] bf16 (transpose + convert)
__global__ __launch_bounds__(256) void transpose_cvt_kernel(const float* in, __bf16* out,
                                                            int R, int C) {
    __shared__ float tile[32][33];
    size_t base = (size_t)blockIdx.z * R * C;
    int r0 = blockIdx.y * 32, c0 = blockIdx.x * 32;
    int tx = threadIdx.x & 31, ty = threadIdx.x >> 5;   // ty 0..7
#pragma unroll
    for (int i = 0; i < 32; i += 8)
        tile[ty + i][tx] = in[base + (size_t)(r0 + ty + i) * C + (c0 + tx)];
    __syncthreads();
#pragma unroll
    for (int i = 0; i < 32; i += 8)
        out[base + (size_t)(c0 + ty + i) * R + (r0 + tx)] = to_bf16(tile[tx][ty + i]);
}

// ---------------- router ----------------
__global__ __launch_bounds__(256) void router_kernel(const float* __restrict__ x,
                                                     const float* __restrict__ gw,
                                                     const float* __restrict__ gb,
                                                     const float* __restrict__ temp,
                                                     int* cnt, int* tok, float* wgt,
                                                     float* psum) {
    int t = blockIdx.x * blockDim.x + threadIdx.x;
    if (t >= NTOK) return;
    float acc[Ee];
#pragma unroll
    for (int e = 0; e < Ee; ++e) acc[e] = gb[e];
    const float* xr = x + (size_t)t * Dd;
    const float4* gw4 = (const float4*)gw;              // [Dd][Ee] -> 2 float4 per row
    for (int j = 0; j < Dd; ++j) {
        float xv = xr[j];
        float4 g0 = gw4[j * 2 + 0];
        float4 g1 = gw4[j * 2 + 1];
        acc[0] += xv * g0.x; acc[1] += xv * g0.y; acc[2] += xv * g0.z; acc[3] += xv * g0.w;
        acc[4] += xv * g1.x; acc[5] += xv * g1.y; acc[6] += xv * g1.z; acc[7] += xv * g1.w;
    }
    float invT = 1.0f / fabsf(temp[0]);
#pragma unroll
    for (int e = 0; e < Ee; ++e) acc[e] *= invT;
    // softmax
    float mx = acc[0];
#pragma unroll
    for (int e = 1; e < Ee; ++e) mx = fmaxf(mx, acc[e]);
    float p[Ee], s = 0.0f;
#pragma unroll
    for (int e = 0; e < Ee; ++e) { p[e] = __expf(acc[e] - mx); s += p[e]; }
    float inv = 1.0f / s;
#pragma unroll
    for (int e = 0; e < Ee; ++e) { p[e] *= inv; atomicAdd(&psum[e], p[e]); }
    // top-2 (strict > keeps lowest index on ties, matching top_k)
    int i1 = 0;
#pragma unroll
    for (int e = 1; e < Ee; ++e) if (p[e] > p[i1]) i1 = e;
    int i2 = (i1 == 0) ? 1 : 0;
#pragma unroll
    for (int e = 0; e < Ee; ++e) if (e != i1 && p[e] > p[i2]) i2 = e;
    float w1v = p[i1], w2v = p[i2], sw = w1v + w2v;
    w1v /= sw; w2v /= sw;
    int s1 = atomicAdd(&cnt[i1], 1);
    if (s1 < CAP) { tok[i1 * CAP + s1] = t; wgt[i1 * CAP + s1] = w1v; }
    int s2 = atomicAdd(&cnt[i2], 1);
    if (s2 < CAP) { tok[i2 * CAP + s2] = t; wgt[i2 * CAP + s2] = w2v; }
}

// ---------------- WMMA GEMM 1: h = relu(gather(x) @ w1 + b1) ----------------
// A: xbf rows gathered by tok   [CAP x Dd]  (bf16, row-major)
// B: w1t [Ee][Hh][Dd]           (bf16, pre-transposed: row = out col, contiguous K)
__global__ __launch_bounds__(256) void gemm1_kernel(const __bf16* __restrict__ xbf,
                                                    const __bf16* __restrict__ w1t,
                                                    const float* __restrict__ b1,
                                                    const int* __restrict__ tok,
                                                    __bf16* __restrict__ hbuf) {
    __shared__ __bf16 smem[2 * TILE_ELEMS];     // double-buffered [A|B], 64 KB
    const int e  = blockIdx.z;
    const int m0 = blockIdx.y * BM;
    const int n0 = blockIdx.x * BN;
    const int tid  = threadIdx.x;
    const int lane = tid & 31;
    const int wave = tid >> 5;
    const int wm = (wave >> 1) * 32;     // wave row offset in tile
    const int wn = (wave & 1) * 64;      // wave col offset in tile
    const int l15 = lane & 15;
    const int hi  = lane >> 4;           // 0 | 1
    const int aoff = hi * 8;             // A frag k split
    const int boff = hi * 8;             // B frag k split

    // global->LDS mapping: 1024 chunks of 8 bf16 per 128x64 tile, 4 per thread
    const int rr = tid >> 3, koff = (tid & 7) * 8;   // rr 0..31
    int   trow[4];
    const __bf16* Ap[4];
    const __bf16* Bp[4];
    unsigned oA[4], oB[4];
#pragma unroll
    for (int i = 0; i < 4; ++i) {
        int r = rr + 32 * i;
        trow[i] = tok[e * CAP + m0 + r];
        Ap[i] = xbf + (size_t)trow[i] * Dd + koff;
        Bp[i] = w1t + ((size_t)e * Hh + (n0 + r)) * Dd + koff;
        oA[i] = (unsigned)((r * BKE + koff) * 2);
        oB[i] = (unsigned)((A_ELEMS + r * BKE + koff) * 2);
    }
    const unsigned lds0 = (unsigned)(size_t)(void*)smem;   // LDS offset = addr[31:0]
    const unsigned lds1 = lds0 + BUF_BYTES;

    f32x8 acc[2][4] = {};

    auto issue = [&](unsigned base, auto offc) {
#pragma unroll
        for (int i = 0; i < 4; ++i) {
            async_copy_b128<decltype(offc)::value>(base + oA[i], Ap[i]);
            async_copy_b128<decltype(offc)::value>(base + oB[i], Bp[i]);
        }
    };
    auto compute = [&](const __bf16* bufA, const __bf16* bufB) {
#pragma unroll
        for (int kk = 0; kk < BKE; kk += 32) {
            AFrag a[2], b[4];
#pragma unroll
            for (int mi = 0; mi < 2; ++mi) {
                const __bf16* p = &bufA[(wm + mi * 16 + l15) * BKE + kk + aoff];
                a[mi].h[0] = *(const bf16x8*)p;
                a[mi].h[1] = *(const bf16x8*)(p + 16);
            }
#pragma unroll
            for (int ni = 0; ni < 4; ++ni) {
                const __bf16* p = &bufB[(wn + ni * 16 + l15) * BKE + kk + boff];
                b[ni].h[0] = *(const bf16x8*)p;
                b[ni].h[1] = *(const bf16x8*)(p + 16);
            }
#pragma unroll
            for (int mi = 0; mi < 2; ++mi)
#pragma unroll
                for (int ni = 0; ni < 4; ++ni)
                    acc[mi][ni] = __builtin_amdgcn_wmma_f32_16x16x32_bf16(
                        false, a[mi].v, false, b[ni].v, (short)0, acc[mi][ni], false, false);
        }
    };

    const __bf16* bufA0 = smem;
    const __bf16* bufB0 = smem + A_ELEMS;
    const __bf16* bufA1 = smem + TILE_ELEMS;
    const __bf16* bufB1 = bufA1 + A_ELEMS;

    const int NST = Dd / BKE;                 // 32 stages, even
    issue(lds0, std::integral_constant<int, 0>{});
    for (int s = 0; s < NST; s += 2) {
        // stage s (buf0): prefetch stage s+1 into buf1 (+BKE elems = +128 B)
        issue(lds1, std::integral_constant<int, BKE * 2>{});
        wait_async8();
        __syncthreads();
        compute(bufA0, bufB0);
        __syncthreads();
        // stage s+1 (buf1): prefetch stage s+2 into buf0 (+2*BKE elems = +256 B)
        if (s + 2 < NST) { issue(lds0, std::integral_constant<int, BKE * 4>{}); wait_async8(); }
        else             { wait_async0(); }
        __syncthreads();
        compute(bufA1, bufB1);
        __syncthreads();
#pragma unroll
        for (int i = 0; i < 4; ++i) { Ap[i] += 2 * BKE; Bp[i] += 2 * BKE; }
    }

    // epilogue: bias + relu -> bf16 h
    const int rbase = m0 + wm + hi * 8;
    const int cbase = n0 + wn + l15;
#pragma unroll
    for (int mi = 0; mi < 2; ++mi)
#pragma unroll
        for (int ni = 0; ni < 4; ++ni) {
            int col = cbase + ni * 16;
            float bias = b1[e * Hh + col];
#pragma unroll
            for (int g = 0; g < 8; ++g) {
                int row = rbase + mi * 16 + g;
                float v = fmaxf(acc[mi][ni][g] + bias, 0.0f);
                hbuf[((size_t)e * CAP + row) * Hh + col] = to_bf16(v);
            }
        }
}

// ---------------- WMMA GEMM 2: out[tok] += (h @ w2 + b2) * w ----------------
__global__ __launch_bounds__(256) void gemm2_kernel(const __bf16* __restrict__ hbuf,
                                                    const __bf16* __restrict__ w2t,
                                                    const float* __restrict__ b2,
                                                    const int* __restrict__ tok,
                                                    const float* __restrict__ wgt,
                                                    float* __restrict__ out) {
    __shared__ __bf16 smem[2 * TILE_ELEMS];
    const int e  = blockIdx.z;
    const int m0 = blockIdx.y * BM;
    const int n0 = blockIdx.x * BN;
    const int tid  = threadIdx.x;
    const int lane = tid & 31;
    const int wave = tid >> 5;
    const int wm = (wave >> 1) * 32;
    const int wn = (wave & 1) * 64;
    const int l15 = lane & 15;
    const int hi  = lane >> 4;
    const int aoff = hi * 8;
    const int boff = hi * 8;

    const int rr = tid >> 3, koff = (tid & 7) * 8;
    const __bf16* Ap[4];
    const __bf16* Bp[4];
    unsigned oA[4], oB[4];
#pragma unroll
    for (int i = 0; i < 4; ++i) {
        int r = rr + 32 * i;
        Ap[i] = hbuf + ((size_t)e * CAP + m0 + r) * Hh + koff;
        Bp[i] = w2t + ((size_t)e * Oo + (n0 + r)) * Hh + koff;
        oA[i] = (unsigned)((r * BKE + koff) * 2);
        oB[i] = (unsigned)((A_ELEMS + r * BKE + koff) * 2);
    }
    const unsigned lds0 = (unsigned)(size_t)(void*)smem;
    const unsigned lds1 = lds0 + BUF_BYTES;

    f32x8 acc[2][4] = {};

    auto issue = [&](unsigned base, auto offc) {
#pragma unroll
        for (int i = 0; i < 4; ++i) {
            async_copy_b128<decltype(offc)::value>(base + oA[i], Ap[i]);
            async_copy_b128<decltype(offc)::value>(base + oB[i], Bp[i]);
        }
    };
    auto compute = [&](const __bf16* bufA, const __bf16* bufB) {
#pragma unroll
        for (int kk = 0; kk < BKE; kk += 32) {
            AFrag a[2], b[4];
#pragma unroll
            for (int mi = 0; mi < 2; ++mi) {
                const __bf16* p = &bufA[(wm + mi * 16 + l15) * BKE + kk + aoff];
                a[mi].h[0] = *(const bf16x8*)p;
                a[mi].h[1] = *(const bf16x8*)(p + 16);
            }
#pragma unroll
            for (int ni = 0; ni < 4; ++ni) {
                const __bf16* p = &bufB[(wn + ni * 16 + l15) * BKE + kk + boff];
                b[ni].h[0] = *(const bf16x8*)p;
                b[ni].h[1] = *(const bf16x8*)(p + 16);
            }
#pragma unroll
            for (int mi = 0; mi < 2; ++mi)
#pragma unroll
                for (int ni = 0; ni < 4; ++ni)
                    acc[mi][ni] = __builtin_amdgcn_wmma_f32_16x16x32_bf16(
                        false, a[mi].v, false, b[ni].v, (short)0, acc[mi][ni], false, false);
        }
    };

    const __bf16* bufA0 = smem;
    const __bf16* bufB0 = smem + A_ELEMS;
    const __bf16* bufA1 = smem + TILE_ELEMS;
    const __bf16* bufB1 = bufA1 + A_ELEMS;

    const int NST = Hh / BKE;
    issue(lds0, std::integral_constant<int, 0>{});
    for (int s = 0; s < NST; s += 2) {
        issue(lds1, std::integral_constant<int, BKE * 2>{});
        wait_async8();
        __syncthreads();
        compute(bufA0, bufB0);
        __syncthreads();
        if (s + 2 < NST) { issue(lds0, std::integral_constant<int, BKE * 4>{}); wait_async8(); }
        else             { wait_async0(); }
        __syncthreads();
        compute(bufA1, bufB1);
        __syncthreads();
#pragma unroll
        for (int i = 0; i < 4; ++i) { Ap[i] += 2 * BKE; Bp[i] += 2 * BKE; }
    }

    // epilogue: bias, scale by routing weight, scatter-add into out
    const int rbase = m0 + wm + hi * 8;
    const int cbase = n0 + wn + l15;
    int   tk[2][8];
    float wv[2][8];
#pragma unroll
    for (int mi = 0; mi < 2; ++mi)
#pragma unroll
        for (int g = 0; g < 8; ++g) {
            int row = rbase + mi * 16 + g;
            tk[mi][g] = tok[e * CAP + row];
            wv[mi][g] = wgt[e * CAP + row];
        }
#pragma unroll
    for (int mi = 0; mi < 2; ++mi)
#pragma unroll
        for (int ni = 0; ni < 4; ++ni) {
            int col = cbase + ni * 16;
            float bias = b2[e * Oo + col];
#pragma unroll
            for (int g = 0; g < 8; ++g) {
                float v = (acc[mi][ni][g] + bias) * wv[mi][g];
                atomicAdd(&out[(size_t)tk[mi][g] * Oo + col], v);
            }
        }
}

// ---------------- aux loss ----------------
__global__ void aux_kernel(const float* psum, const int* cnt, float* out_aux) {
    if (threadIdx.x == 0) {
        float bal = 0.0f, mean = 0.0f;
        float imp[Ee];
        for (int e = 0; e < Ee; ++e) {
            imp[e] = psum[e];
            float ap = imp[e] / (float)NTOK;
            float au = (float)cnt[e] / (float)NTOK;
            bal += ap * au;
            mean += imp[e];
        }
        bal *= (float)Ee;
        mean /= (float)Ee;
        float var = 0.0f;
        for (int e = 0; e < Ee; ++e) { float d = imp[e] - mean; var += d * d; }
        var /= (float)(Ee - 1);
        float cv = sqrtf(var) / (mean + 1e-10f);
        *out_aux = bal + 0.01f * cv;
    }
}

// ---------------- launch ----------------
#include <type_traits>
extern "C" void kernel_launch(void* const* d_in, const int* in_sizes, int n_in,
                              void* d_out, int out_size, void* d_ws, size_t ws_size,
                              hipStream_t stream) {
    const float* x      = (const float*)d_in[0];
    const float* gate_w = (const float*)d_in[1];
    const float* gate_b = (const float*)d_in[2];
    const float* temp   = (const float*)d_in[3];
    const float* w1     = (const float*)d_in[4];
    const float* b1     = (const float*)d_in[5];
    const float* w2     = (const float*)d_in[6];
    const float* b2     = (const float*)d_in[7];
    float* out = (float*)d_out;

    char* ws = (char*)d_ws;
    size_t o = 0;
    int*    cnt  = (int*)(ws + o);   o += 256;
    float*  psum = (float*)(ws + o); o += 256;
    int*    tok  = (int*)(ws + o);   o += (size_t)Ee * CAP * 4;
    float*  wgt  = (float*)(ws + o); o += (size_t)Ee * CAP * 4;
    __bf16* xbf  = (__bf16*)(ws + o); o += (size_t)NTOK * Dd * 2;
    __bf16* w1t  = (__bf16*)(ws + o); o += (size_t)Ee * Dd * Hh * 2;
    __bf16* w2t  = (__bf16*)(ws + o); o += (size_t)Ee * Hh * Oo * 2;
    __bf16* hbuf = (__bf16*)(ws + o); o += (size_t)Ee * CAP * Hh * 2;
    (void)ws_size; (void)in_sizes; (void)n_in; (void)out_size;

    // 1. init metadata
    init_meta_kernel<<<(Ee * CAP + 255) / 256, 256, 0, stream>>>(cnt, psum, tok, wgt);
    // 2. residual init: out = x
    out_init_kernel<<<4096, 256, 0, stream>>>((float4*)out, (const float4*)x,
                                              (int)((size_t)NTOK * Oo / 4));
    // 3. conversions
    cvt_x_kernel<<<4096, 256, 0, stream>>>(x, xbf, NTOK * Dd);
    transpose_cvt_kernel<<<dim3(Hh / 32, Dd / 32, Ee), 256, 0, stream>>>(w1, w1t, Dd, Hh);
    transpose_cvt_kernel<<<dim3(Oo / 32, Hh / 32, Ee), 256, 0, stream>>>(w2, w2t, Hh, Oo);
    // 4. router
    router_kernel<<<NTOK / 256, 256, 0, stream>>>(x, gate_w, gate_b, temp,
                                                  cnt, tok, wgt, psum);
    // 5. expert GEMMs
    gemm1_kernel<<<dim3(Hh / BN, CAP / BM, Ee), 256, 0, stream>>>(xbf, w1t, b1, tok, hbuf);
    gemm2_kernel<<<dim3(Oo / BN, CAP / BM, Ee), 256, 0, stream>>>(hbuf, w2t, b2, tok, wgt, out);
    // 6. aux loss scalar
    aux_kernel<<<1, 32, 0, stream>>>(psum, cnt, out + (size_t)NTOK * Oo);
}
